// GPT_1365799600156
// MI455X (gfx1250) — matrix operand order repaired and checked by reading, compiled, not run on hardware
//
#include <hip/hip_runtime.h>

typedef __attribute__((ext_vector_type(16))) __bf16 v16bf;
typedef __attribute__((ext_vector_type(8)))  __bf16 v8bf;
typedef __attribute__((ext_vector_type(8)))  float  v8f;

constexpr int Bc   = 2;
constexpr int Tc   = 2048;
constexpr int Cc   = 2048;
constexpr int NH   = 16;
constexpr int G    = 4;
constexpr int HS   = 128;
constexpr int QPK  = NH / G;          // 4
constexpr int WIN  = 512;
constexpr int QKVC = (NH + 2 * G) * HS; // 3072
constexpr int NHHS = NH * HS;           // 2048
constexpr float SCALE = 0.08838834764831845f; // 1/sqrt(128)

// ---------------------------------------------------------------------------
// Fragment load helpers (CDNA5 WMMA bf16 16x16x32 layouts, ISA 7.12.2)
// A (16x32, row-major [M,K]): lane m=l%16; lanes<16 hold K 0..7 & 16..23,
// lanes>=16 hold K 8..15 & 24..31  -> two 16B loads at +0 and +16 elements.
// B (as B^T row-major [N,K]):  lane n=l%16; same K order as A
//                              -> two 16B loads at +0 and +16 elements.
// ---------------------------------------------------------------------------
static __device__ __forceinline__ v16bf ld_a16(const __bf16* p) {
  v8bf lo = *reinterpret_cast<const v8bf*>(p);
  v8bf hi = *reinterpret_cast<const v8bf*>(p + 16);
  return __builtin_shufflevector(lo, hi, 0,1,2,3,4,5,6,7,8,9,10,11,12,13,14,15);
}
static __device__ __forceinline__ v16bf ld_b16(const __bf16* p) {
  v8bf lo = *reinterpret_cast<const v8bf*>(p);
  v8bf hi = *reinterpret_cast<const v8bf*>(p + 16);
  return __builtin_shufflevector(lo, hi, 0,1,2,3,4,5,6,7,8,9,10,11,12,13,14,15);
}
static __device__ __forceinline__ v8f wmma_bf(v16bf a, v16bf b, v8f c) {
  return __builtin_amdgcn_wmma_f32_16x16x32_bf16(false, a, false, b,
                                                 (short)0, c, false, false);
}
// Async copy 16B global -> LDS (tracked by ASYNCcnt).  lds = byte offset
// (low 32 bits of a generic shared pointer ARE the LDS offset on amdgcn).
static __device__ __forceinline__ void async_ld_b128(const __bf16* g,
                                                     unsigned lds) {
  asm volatile("global_load_async_to_lds_b128 %0, %1, off"
               :: "v"(lds), "v"(g) : "memory");
}
static __device__ __forceinline__ void wait_async0() {
  asm volatile("s_wait_asynccnt 0x0" ::: "memory");
}

// ---------------------------------------------------------------------------
// Prep kernels
// ---------------------------------------------------------------------------
__global__ void cvt_f32_bf16(const float* __restrict__ in,
                             __bf16* __restrict__ out, int n) {
  int i = blockIdx.x * blockDim.x + threadIdx.x;
  if (i < n) out[i] = (__bf16)in[i];
}

// in [R, N] (row-major f32) -> out [N, R] (row-major bf16)  == B^T layout
__global__ void transpose_bf16(const float* __restrict__ in,
                               __bf16* __restrict__ out, int R, int N) {
  int idx = blockIdx.x * blockDim.x + threadIdx.x;
  if (idx >= R * N) return;
  int n = idx % N;
  int r = idx / N;
  out[(size_t)n * R + r] = (__bf16)in[(size_t)r * N + n];
}

// ---------------------------------------------------------------------------
// bf16 WMMA GEMM: C[M,N] (f32) = A[M,K] (bf16 row-major) * Bt[N,K]^T
// Block = 4 waves; tile 64(M) x 128(N); each wave owns a 16x128 strip.
// B tile (128x32 per K-step) is staged into LDS with async-to-LDS copies,
// double buffered: loads for step s+1 overlap the 8 WMMAs of step s.
// B fragments are loaded into distinct registers before the WMMA burst so
// the scheduler can clause the ds_loads and use partial dscnt waits.
// ---------------------------------------------------------------------------
__global__ __launch_bounds__(128) void gemm_bf16_wmma(
    const __bf16* __restrict__ A, const __bf16* __restrict__ Bt,
    float* __restrict__ C, int M, int N, int K) {
  __shared__ __bf16 bst[2][128][32];   // 2 x 8KB

  const int tid  = threadIdx.x;
  const int lane = tid & 31;
  const int wave = tid >> 5;
  const int half = lane >> 4;   // 0/1
  const int l16  = lane & 15;
  const int m0 = blockIdx.y * 64 + wave * 16;
  const int n0 = blockIdx.x * 128;

  v8f acc[8];
#pragma unroll
  for (int nt = 0; nt < 8; ++nt) acc[nt] = v8f{};

  const __bf16* arow = A + (size_t)(m0 + l16) * K;
  const int NS = K / 32;

  // cooperative async stage of the 128x32 B tile for K-step s
  auto issue = [&](int s) {
    const int kk = s * 32;
    const int bb = s & 1;
#pragma unroll
    for (int c = 0; c < 4; ++c) {
      int chunk = tid * 4 + c;        // 0..511 chunks of 16B
      int row   = chunk >> 2;         // 0..127
      int col   = (chunk & 3) * 8;    // element offset 0/8/16/24
      const __bf16* g = Bt + (size_t)(n0 + row) * K + kk + col;
      async_ld_b128(g, (unsigned)(uintptr_t)&bst[bb][row][col]);
    }
  };

  issue(0);
  for (int s = 0; s < NS; ++s) {
    wait_async0();            // own async writes for buf s&1 have landed
    __syncthreads();          // ... and everyone else's too; everyone is done
                              // reading buf (s+1)&1 from step s-1
    if (s + 1 < NS) issue(s + 1);

    v16bf af = ld_a16(arow + s * 32 + half * 8);
    const int bb = s & 1;
    v16bf bfr[8];
#pragma unroll
    for (int nt = 0; nt < 8; ++nt)
      bfr[nt] = ld_b16(&bst[bb][nt * 16 + l16][half * 8]);
#pragma unroll
    for (int nt = 0; nt < 8; ++nt)
      acc[nt] = wmma_bf(af, bfr[nt], acc[nt]);
  }
#pragma unroll
  for (int nt = 0; nt < 8; ++nt) {
#pragma unroll
    for (int v = 0; v < 8; ++v) {
      int row = m0 + half * 8 + v;    // D layout: m = v + 8*(lane/16)
      C[(size_t)row * N + n0 + nt * 16 + l16] = acc[nt][v];
    }
  }
}

// ---------------------------------------------------------------------------
// causal conv1d (K=4) + SiLU + RoPE for Q and K heads.
// qkv: [B*T, QKVC] f32.  out: [B, H, T, HS] bf16 (head-major for attention).
// Column of channel (h, hs):  (g*(QPK+2) + sub)*HS + hs,
//   Q: g=h/QPK, sub=h%QPK   (perG=QPK, base=0)
//   K: g=h,     sub=QPK     (perG=1,   base=QPK)
// Each thread handles the RoPE pair (d, d+64).
// ---------------------------------------------------------------------------
__global__ void conv_silu_rope(const float* __restrict__ qkv,
                               const float* __restrict__ w, // [H*HS, 4]
                               const float* __restrict__ cosp,
                               const float* __restrict__ sinp,
                               __bf16* __restrict__ out, int H, int perG,
                               int base, int total) {
  int idx = blockIdx.x * blockDim.x + threadIdx.x;
  if (idx >= total) return;
  int d = idx & 63;
  int h = (idx >> 6) % H;
  int t = (idx / (64 * H)) % Tc;
  int b = idx / (64 * H * Tc);
  int g = h / perG;
  int sub = base + (h % perG);
  int col0 = (g * (QPK + 2) + sub) * HS + d;

  float r[2];
#pragma unroll
  for (int p = 0; p < 2; ++p) {
    int ch = h * HS + d + p * 64;
    int col = col0 + p * 64;
    float s = 0.f;
#pragma unroll
    for (int k = 0; k < 4; ++k) {
      int ti = t - 3 + k;
      if (ti >= 0)
        s += w[ch * 4 + k] * qkv[(size_t)(b * Tc + ti) * QKVC + col];
    }
    r[p] = s / (1.f + __expf(-s)); // SiLU
  }
  float cs = cosp[t * 64 + d];
  float sn = sinp[t * 64 + d];
  size_t ob = (((size_t)b * H + h) * Tc + t) * HS + d;
  out[ob]      = (__bf16)(r[0] * cs - r[1] * sn);
  out[ob + 64] = (__bf16)(r[0] * sn + r[1] * cs);
}

// V: conv + SiLU, no RoPE; stored TRANSPOSED [B, G, HS, T] so the P*V
// B-fragment is a contiguous K-run per lane.
__global__ void conv_silu_v(const float* __restrict__ qkv,
                            const float* __restrict__ w, // [G*HS, 4]
                            __bf16* __restrict__ vt, int total) {
  int idx = blockIdx.x * blockDim.x + threadIdx.x;
  if (idx >= total) return;
  int d = idx & (HS - 1);
  int g = (idx >> 7) % G;
  int t = (idx / (HS * G)) % Tc;
  int b = idx / (HS * G * Tc);
  int col = (g * (QPK + 2) + (QPK + 1)) * HS + d;
  int ch = g * HS + d;
  float s = 0.f;
#pragma unroll
  for (int k = 0; k < 4; ++k) {
    int ti = t - 3 + k;
    if (ti >= 0) s += w[ch * 4 + k] * qkv[(size_t)(b * Tc + ti) * QKVC + col];
  }
  s = s / (1.f + __expf(-s));
  vt[(((size_t)b * G + g) * HS + d) * Tc + t] = (__bf16)s;
}

// ---------------------------------------------------------------------------
// Flash-style sliding-window attention. One wave per (b, h, 16-row q tile).
// S = Q K^T via WMMA (K=HS chained), online softmax (row stats replicated
// across each 16-lane half-group), P staged through LDS to convert from
// C-layout to A-layout, then O += P V via WMMA against transposed V.
// ---------------------------------------------------------------------------
__global__ __launch_bounds__(32) void attn_wmma(
    const __bf16* __restrict__ qb,  // [B, NH, T, HS]
    const __bf16* __restrict__ kb,  // [B, G,  T, HS]
    const __bf16* __restrict__ vt,  // [B, G, HS, T]
    __bf16* __restrict__ yb) {      // [B*T, NH*HS]
  __shared__ __bf16 pl[16][32];

  const int lane = threadIdx.x;
  const int half = lane >> 4;
  const int l16  = lane & 15;
  constexpr int TQ = Tc / 16;
  const int qt = blockIdx.x % TQ;
  const int h  = (blockIdx.x / TQ) % NH;
  const int b  = blockIdx.x / (TQ * NH);
  const int g  = h / QPK;
  const int i0 = qt * 16;

  const __bf16* qp = qb + ((size_t)b * NH + h) * Tc * HS;
  const __bf16* kp = kb + ((size_t)b * G + g) * Tc * HS;
  const __bf16* vp = vt + ((size_t)b * G + g) * HS * Tc;

  // Preload the full 16x128 Q tile as four A fragments.
  v16bf qf[4];
#pragma unroll
  for (int kd = 0; kd < 4; ++kd)
    qf[kd] = ld_a16(qp + (size_t)(i0 + l16) * HS + kd * 32 + half * 8);

  v8f o[8];
  float rmax[8], rsum[8];
#pragma unroll
  for (int v = 0; v < 8; ++v) {
    o[v] = v8f{};
    rmax[v] = -1e30f;
    rsum[v] = 0.f;
  }

  int s_lo = i0 - (WIN - 1);
  if (s_lo < 0) s_lo = 0;
  s_lo &= ~31;

  for (int s0 = s_lo; s0 <= i0 + 15; s0 += 32) {
    float pv[2][8];
    // ---- scores for two 16-wide j tiles ----
#pragma unroll
    for (int jt = 0; jt < 2; ++jt) {
      v16bf kfr[4];
#pragma unroll
      for (int kd = 0; kd < 4; ++kd)
        kfr[kd] = ld_b16(kp + (size_t)(s0 + jt * 16 + l16) * HS + kd * 32 +
                         half * 8);
      v8f sa = v8f{};
#pragma unroll
      for (int kd = 0; kd < 4; ++kd) sa = wmma_bf(qf[kd], kfr[kd], sa);
#pragma unroll
      for (int v = 0; v < 8; ++v) {
        int i = i0 + half * 8 + v;
        int j = s0 + jt * 16 + l16;
        bool ok = (j <= i) && ((i - j) < WIN);
        pv[jt][v] = ok ? sa[v] * SCALE : -1e30f;
      }
    }
    // ---- online softmax update (rows live across 16-lane half-groups) ----
#pragma unroll
    for (int v = 0; v < 8; ++v) {
      float m = fmaxf(pv[0][v], pv[1][v]);
      m = fmaxf(m, __shfl_xor(m, 1, 32));
      m = fmaxf(m, __shfl_xor(m, 2, 32));
      m = fmaxf(m, __shfl_xor(m, 4, 32));
      m = fmaxf(m, __shfl_xor(m, 8, 32));
      float nm = fmaxf(rmax[v], m);
      float f = __expf(rmax[v] - nm);
      rmax[v] = nm;
      float p0 = __expf(pv[0][v] - nm);
      float p1 = __expf(pv[1][v] - nm);
      pv[0][v] = p0;
      pv[1][v] = p1;
      float ps = p0 + p1;
      ps += __shfl_xor(ps, 1, 32);
      ps += __shfl_xor(ps, 2, 32);
      ps += __shfl_xor(ps, 4, 32);
      ps += __shfl_xor(ps, 8, 32);
      rsum[v] = rsum[v] * f + ps;
#pragma unroll
      for (int dt = 0; dt < 8; ++dt) o[dt][v] *= f;
    }
    // ---- stage P through LDS: C-layout -> A-layout ----
    __syncthreads();
#pragma unroll
    for (int v = 0; v < 8; ++v) {
      pl[half * 8 + v][l16]      = (__bf16)pv[0][v];
      pl[half * 8 + v][16 + l16] = (__bf16)pv[1][v];
    }
    __syncthreads();
    v16bf pa = ld_a16(&pl[l16][half * 8]);
    // ---- O += P * V  (8 d-tiles of 16) ----
    v16bf vfr[8];
#pragma unroll
    for (int dt = 0; dt < 8; ++dt)
      vfr[dt] = ld_b16(vp + (size_t)(dt * 16 + l16) * Tc + s0 + half * 8);
#pragma unroll
    for (int dt = 0; dt < 8; ++dt) o[dt] = wmma_bf(pa, vfr[dt], o[dt]);
  }
  // ---- normalize + store y (bf16, row-major [B*T, NH*HS]) ----
#pragma unroll
  for (int v = 0; v < 8; ++v) {
    float inv = 1.0f / rsum[v];
    int tr = i0 + half * 8 + v;
#pragma unroll
    for (int dt = 0; dt < 8; ++dt) {
      yb[(size_t)(b * Tc + tr) * NHHS + h * HS + dt * 16 + l16] =
          (__bf16)(o[dt][v] * inv);
    }
  }
}

// ---------------------------------------------------------------------------
// Host-side orchestration
// ---------------------------------------------------------------------------
static inline size_t alignUp(size_t x) { return (x + 255) & ~size_t(255); }

extern "C" void kernel_launch(void* const* d_in, const int* in_sizes, int n_in,
                              void* d_out, int out_size, void* d_ws,
                              size_t ws_size, hipStream_t stream) {
  (void)in_sizes; (void)n_in; (void)out_size; (void)ws_size;

  const float* x    = (const float*)d_in[0]; // [B,T,C]
  const float* Wqkv = (const float*)d_in[1]; // [C, QKVC]
  const float* Wpro = (const float*)d_in[2]; // [NH*HS, C]
  const float* qcw  = (const float*)d_in[3]; // [NH*HS, 4]
  const float* kcw  = (const float*)d_in[4]; // [G*HS, 4]
  const float* vcw  = (const float*)d_in[5]; // [G*HS, 4]
  const float* cosp = (const float*)d_in[6]; // [T, 64]
  const float* sinp = (const float*)d_in[7]; // [T, 64]
  float* out = (float*)d_out;                // [B,T,C] f32

  const size_t BT = (size_t)Bc * Tc;         // 4096
  uint8_t* ws = (uint8_t*)d_ws;
  size_t off = 0;
  __bf16* xb = (__bf16*)(ws + off);          // also reused as y_bf
  off = alignUp(off + BT * Cc * sizeof(__bf16));
  __bf16* wqkvt = (__bf16*)(ws + off);       // [QKVC, C]
  off = alignUp(off + (size_t)QKVC * Cc * sizeof(__bf16));
  float* qkv = (float*)(ws + off);           // [B*T, QKVC] f32
  __bf16* wprot = (__bf16*)(ws + off);       // aliases qkv (used after conv)
  off = alignUp(off + BT * QKVC * sizeof(float));
  __bf16* q_bf = (__bf16*)(ws + off);        // [B,NH,T,HS]
  off = alignUp(off + (size_t)Bc * NH * Tc * HS * sizeof(__bf16));
  __bf16* k_bf = (__bf16*)(ws + off);        // [B,G,T,HS]
  off = alignUp(off + (size_t)Bc * G * Tc * HS * sizeof(__bf16));
  __bf16* v_t = (__bf16*)(ws + off);         // [B,G,HS,T]
  off = alignUp(off + (size_t)Bc * G * HS * Tc * sizeof(__bf16));
  __bf16* y_bf = xb;                         // xb dead after GEMM1

  // 1) x -> bf16
  {
    int n = (int)(BT * Cc);
    cvt_f32_bf16<<<(n + 255) / 256, 256, 0, stream>>>(x, xb, n);
  }
  // 2) Wqkv^T -> bf16  (N-major for B-fragments)
  {
    int n = Cc * QKVC;
    transpose_bf16<<<(n + 255) / 256, 256, 0, stream>>>(Wqkv, wqkvt, Cc, QKVC);
  }
  // 3) qkv = x @ Wqkv   (WMMA + async-LDS double buffering)
  {
    dim3 grid(QKVC / 128, (int)(BT / 64));
    gemm_bf16_wmma<<<grid, 128, 0, stream>>>(xb, wqkvt, qkv, (int)BT, QKVC, Cc);
  }
  // 4) conv+SiLU+RoPE for Q and K; conv+SiLU for V (transposed)
  {
    int n = (int)(BT * NH * 64);
    conv_silu_rope<<<(n + 255) / 256, 256, 0, stream>>>(qkv, qcw, cosp, sinp,
                                                        q_bf, NH, QPK, 0, n);
  }
  {
    int n = (int)(BT * G * 64);
    conv_silu_rope<<<(n + 255) / 256, 256, 0, stream>>>(qkv, kcw, cosp, sinp,
                                                        k_bf, G, 1, QPK, n);
  }
  {
    int n = (int)(BT * G * HS);
    conv_silu_v<<<(n + 255) / 256, 256, 0, stream>>>(qkv, vcw, v_t, n);
  }
  // 5) Wproj^T -> bf16 (into the now-dead qkv area)
  {
    int n = NHHS * Cc;
    transpose_bf16<<<(n + 255) / 256, 256, 0, stream>>>(Wpro, wprot, NHHS, Cc);
  }
  // 6) flash attention -> y_bf (reuses xb region)
  {
    dim3 grid(Bc * NH * (Tc / 16));
    attn_wmma<<<grid, 32, 0, stream>>>(q_bf, k_bf, v_t, y_bf);
  }
  // 7) out = y @ Wproj  (WMMA, f32 result)
  {
    dim3 grid(Cc / 128, (int)(BT / 64));
    gemm_bf16_wmma<<<grid, 128, 0, stream>>>(y_bf, wprot, out, (int)BT, Cc,
                                             NHHS);
  }
}
